// RSSM_28269474742840
// MI455X (gfx1250) — matrix operand order repaired and checked by reading, not compiled, not run on hardware
//
#include <hip/hip_runtime.h>
#include <hip/hip_bf16.h>

// ---------------------------------------------------------------------------
// RSSM observe scan for MI455X (gfx1250, wave32, WMMA bf16 K=32, f32 accum).
// One persistent kernel: block = 16 batch rows for all T steps (recurrences
// are independent per batch row -> no inter-block sync needed).
// ---------------------------------------------------------------------------

typedef __attribute__((ext_vector_type(16))) __bf16 v16bf;
typedef __attribute__((ext_vector_type(8)))  __bf16 v8bf;
typedef __attribute__((ext_vector_type(8)))  float  v8f;

#define B_   512
#define T_   64
#define E_   1024
#define A_   32
#define S_   30
#define D_   200
#define H_   200
#define SS_  230
#define OUT_ 380          // 6*S + D
#define MIN_STD_     0.1f
#define UPDATE_BIAS_ (-1.0f)

// padded GEMM dims (K padded to mult of 32, N padded to mult of 16)
#define K1P 320   // 292 (S+SS+A)
#define N1P 208   // 200
#define K2P 416   // 400 (H+D)
#define N2P 608   // 600
#define K3P 224   // 200
#define N3P 208   // 200
#define N4P 64    // 60
#define K5P 1248  // 1224 (D+E)
#define N5P 208   // 200
#define N6P 64    // 60

// ---------------------------------------------------------------------------
// Branchless fast activations (v_exp_f32 / v_log_f32 / fast divide).
// GEMM inputs are bf16 already, so ocml-accurate transcendentals buy nothing.
// ---------------------------------------------------------------------------
__device__ __forceinline__ float sigmoidf_(float x)
{
    return __fdividef(1.0f, 1.0f + __expf(-x));
}
__device__ __forceinline__ float tanhf_(float x)
{
    // tanh(x) = 1 - 2/(exp(2x)+1); exact limits at +-inf, no branches
    float e = __expf(2.0f * x);
    return 1.0f - __fdividef(2.0f, e + 1.0f);
}
__device__ __forceinline__ float eluf_(float x)
{
    // x>0:  x + exp(0)-1 = x ;  x<=0: 0 + exp(x)-1
    return fmaxf(x, 0.0f) + __expf(fminf(x, 0.0f)) - 1.0f;
}
__device__ __forceinline__ float softplusf_(float x)
{
    return fmaxf(x, 0.0f) + __logf(1.0f + __expf(-fabsf(x)));
}

// A fragment: 16x32 bf16 tile from LDS row-major [16][KP].
// Contiguous K order used here (B matches): lane L: row = L%16; elems 0..15 -> K = kc*32 + (L/16)*16 + e.
// B fragment: weights pre-transposed to [Npad][KP] bf16 so lane L reads
// 16 contiguous K for n = nt*16 + L%16, K-half selected by L/16.
__device__ __forceinline__ void gemm_tile(const __bf16* __restrict__ sA, int KP,
                                          const __bf16* __restrict__ wt,
                                          int kchunks, int nt, int lane, v8f& acc)
{
    const int nl = lane & 15;
    const int kh = lane >> 4;
    const __bf16* a0 = sA + nl * KP + kh * 16;
    const __bf16* b0 = wt + (size_t)(nt * 16 + nl) * KP + kh * 16;
    for (int kc = 0; kc < kchunks; ++kc) {
        v8bf lo = *(const v8bf*)(a0 + kc * 32);
        v8bf hi = *(const v8bf*)(a0 + kc * 32 + 8);
        v16bf a = __builtin_shufflevector(lo, hi, 0,1,2,3,4,5,6,7,8,9,10,11,12,13,14,15);
        v16bf b = *(const v16bf*)(b0 + kc * 32);
        acc = __builtin_amdgcn_wmma_f32_16x16x32_bf16(false, a, false, b,
                                                      (short)0, acc, false, false);
    }
}

// C/D layout: lane L: n = L%16, m = (L/16)*8 + r  (r = accumulator index 0..7)
// bv = bias[n], preloaded BEFORE the WMMA chain so its latency is hidden.
__device__ __forceinline__ void store_elu_bf16(const v8f& acc, float bv,
                                               __bf16* __restrict__ dst, int stride,
                                               int n, int nvalid, int lane)
{
    if (n >= nvalid) return;
    int mb = (lane >> 4) * 8;
    #pragma unroll
    for (int r = 0; r < 8; ++r)
        dst[(mb + r) * stride + n] = (__bf16)eluf_(acc[r] + bv);
}

__device__ __forceinline__ void store_f32(const v8f& acc, float bv,
                                          float* __restrict__ dst, int stride,
                                          int n, int nvalid, int lane)
{
    if (n >= nvalid) return;
    int mb = (lane >> 4) * 8;
    #pragma unroll
    for (int r = 0; r < 8; ++r)
        dst[(mb + r) * stride + n] = acc[r] + bv;
}

// ---------------------------------------------------------------------------
// Weight prep: transpose f32 [K][N] -> padded bf16 [NP][KP] (WMMA-B friendly)
// ---------------------------------------------------------------------------
__global__ void prep_weight_kernel(const float* __restrict__ src, __bf16* __restrict__ dst,
                                   int K, int N, int KP, int NP)
{
    int idx = blockIdx.x * blockDim.x + threadIdx.x;
    if (idx >= NP * KP) return;
    int n = idx / KP, k = idx % KP;
    float v = (n < N && k < K) ? src[(size_t)k * N + n] : 0.0f;
    dst[idx] = (__bf16)v;
}

// ---------------------------------------------------------------------------
// Main persistent scan kernel. grid = B/16 = 32 blocks, 256 threads (8 waves).
// ---------------------------------------------------------------------------
__global__ __launch_bounds__(256)
void rssm_observe_kernel(const float* __restrict__ embed,   // [B,T,E]
                         const float* __restrict__ action,  // [B,T,A]
                         const float* __restrict__ context, // [B,T,SS]
                         const float* __restrict__ noise_p, // [T,B,S]
                         const float* __restrict__ noise_o, // [T,B,S]
                         const __bf16* __restrict__ wt_inp, const float* __restrict__ b_inp,
                         const __bf16* __restrict__ wt_gru, const float* __restrict__ b_gru,
                         const __bf16* __restrict__ wt_img, const float* __restrict__ b_img,
                         const __bf16* __restrict__ wt_obs, const float* __restrict__ b_obs,
                         const __bf16* __restrict__ wt_ims, const float* __restrict__ b_ims,
                         const __bf16* __restrict__ wt_ost, const float* __restrict__ b_ost,
                         float* __restrict__ out)            // [B,T,380]
{
    __shared__ __align__(16) __bf16 sA1 [16 * K1P]; // [stoch | ctx | act | 0]
    __shared__ __align__(16) __bf16 sX  [16 * K2P]; // [x | deter_bf | 0]
    __shared__ __align__(16) __bf16 sDbf[16 * K3P]; // deter bf16
    __shared__ __align__(16) __bf16 sH  [16 * K3P]; // img hidden bf16
    __shared__ __align__(16) __bf16 sHo [16 * K3P]; // obs hidden bf16
    __shared__ __align__(16) __bf16 sXo [16 * K5P]; // [deter_bf | emb | 0]
    __shared__ float sParts[16 * N2P];              // GRU pre-activations f32
    __shared__ float sDeter[16 * D_];               // fp32 deter carry
    __shared__ float sStat [16 * N4P];              // prior stats f32
    __shared__ float sStat2[16 * N4P];              // posterior stats f32

    const int tid   = threadIdx.x;
    const int lane  = tid & 31;
    const int wave  = tid >> 5;
    const int nwave = blockDim.x >> 5;
    const int nl    = lane & 15;
    const int m0    = blockIdx.x * 16;

    // zero state + all K/N pads once (pads are never re-written afterwards)
    for (int i = tid; i < 16 * K1P; i += blockDim.x) sA1[i] = (__bf16)0.0f;
    for (int i = tid; i < 16 * K2P; i += blockDim.x) sX[i]  = (__bf16)0.0f;
    for (int i = tid; i < 16 * K3P; i += blockDim.x) { sDbf[i] = (__bf16)0.0f; sH[i] = (__bf16)0.0f; sHo[i] = (__bf16)0.0f; }
    for (int i = tid; i < 16 * K5P; i += blockDim.x) sXo[i] = (__bf16)0.0f;
    for (int i = tid; i < 16 * D_;  i += blockDim.x) sDeter[i] = 0.0f;
    __syncthreads();

    for (int t = 0; t < T_; ++t) {
        // ---- phase 0: stage ctx/act into sA1 and emb into sXo (bf16) ------
        for (int i = tid; i < 16 * SS_; i += blockDim.x) {
            int r = i / SS_, k = i % SS_;
            sA1[r * K1P + S_ + k] =
                (__bf16)context[(size_t)(m0 + r) * (T_ * SS_) + (size_t)t * SS_ + k];
        }
        for (int i = tid; i < 16 * A_; i += blockDim.x) {
            int r = i / A_, k = i % A_;
            sA1[r * K1P + S_ + SS_ + k] =
                (__bf16)action[(size_t)(m0 + r) * (T_ * A_) + (size_t)t * A_ + k];
        }
        for (int i = tid; i < 16 * E_; i += blockDim.x) {
            int r = i / E_, k = i % E_;
            const float* ep = embed + (size_t)(m0 + r) * (T_ * E_) + (size_t)t * E_ + k;
            sXo[r * K5P + D_ + k] = (__bf16)(*ep);
            if (t + 1 < T_) __builtin_prefetch(ep + E_, 0, 0);   // next step's embed
        }
        __syncthreads();

        // ---- GEMM1: x = elu([stoch,ctx,act] @ w_inp + b) -> sX[:,0:200] ---
        for (int nt = wave; nt < N1P / 16; nt += nwave) {
            int n = nt * 16 + nl;
            float bv = (n < H_) ? b_inp[n] : 0.0f;          // issued pre-WMMA
            v8f acc; for (int i = 0; i < 8; ++i) acc[i] = 0.0f;
            gemm_tile(sA1, K1P, wt_inp, K1P / 32, nt, lane, acc);
            store_elu_bf16(acc, bv, sX, K2P, n, H_, lane);
        }
        __syncthreads();

        // ---- GEMM2: parts = [x,deter] @ w_gru + b -> sParts (f32) ---------
        for (int nt = wave; nt < N2P / 16; nt += nwave) {
            int n = nt * 16 + nl;
            float bv = (n < 3 * D_) ? b_gru[n] : 0.0f;
            v8f acc; for (int i = 0; i < 8; ++i) acc[i] = 0.0f;
            gemm_tile(sX, K2P, wt_gru, K2P / 32, nt, lane, acc);
            store_f32(acc, bv, sParts, N2P, n, 3 * D_, lane);
        }
        __syncthreads();

        // ---- GRU elementwise update (fp32 carry) --------------------------
        for (int i = tid; i < 16 * D_; i += blockDim.x) {
            int r = i / D_, d = i % D_;
            float reset  = sigmoidf_(sParts[r * N2P + d]);
            float cand   = tanhf_(reset * sParts[r * N2P + D_ + d]);
            float update = sigmoidf_(sParts[r * N2P + 2 * D_ + d] + UPDATE_BIAS_);
            float dn = update * cand + (1.0f - update) * sDeter[i];
            sDeter[i] = dn;
            __bf16 db = (__bf16)dn;
            sX [r * K2P + H_ + d] = db;
            sXo[r * K5P + d]      = db;
            sDbf[r * K3P + d]     = db;
            out[(size_t)(m0 + r) * (T_ * OUT_) + (size_t)t * OUT_ + 6 * S_ + d] = dn;
        }
        __syncthreads();

        // ---- GEMM5 (obs hidden, heavy K=1248) || GEMM3 (img hidden) -------
        for (int j = wave; j < (N5P / 16) + (N3P / 16); j += nwave) {
            v8f acc; for (int i = 0; i < 8; ++i) acc[i] = 0.0f;
            if (j < N5P / 16) {
                int n = j * 16 + nl;
                float bv = (n < H_) ? b_obs[n] : 0.0f;
                gemm_tile(sXo, K5P, wt_obs, K5P / 32, j, lane, acc);
                store_elu_bf16(acc, bv, sHo, K3P, n, H_, lane);
            } else {
                int nt = j - N5P / 16;
                int n = nt * 16 + nl;
                float bv = (n < H_) ? b_img[n] : 0.0f;
                gemm_tile(sDbf, K3P, wt_img, K3P / 32, nt, lane, acc);
                store_elu_bf16(acc, bv, sH, K3P, n, H_, lane);
            }
        }
        __syncthreads();

        // ---- GEMM4 (prior stats) || GEMM6 (posterior stats) ---------------
        for (int j = wave; j < (N4P / 16) + (N6P / 16); j += nwave) {
            v8f acc; for (int i = 0; i < 8; ++i) acc[i] = 0.0f;
            if (j < N4P / 16) {
                int n = j * 16 + nl;
                float bv = (n < 2 * S_) ? b_ims[n] : 0.0f;
                gemm_tile(sH, K3P, wt_ims, K3P / 32, j, lane, acc);
                store_f32(acc, bv, sStat, N4P, n, 2 * S_, lane);
            } else {
                int nt = j - N4P / 16;
                int n = nt * 16 + nl;
                float bv = (n < 2 * S_) ? b_ost[n] : 0.0f;
                gemm_tile(sHo, K3P, wt_ost, K3P / 32, nt, lane, acc);
                store_f32(acc, bv, sStat2, N4P, n, 2 * S_, lane);
            }
        }
        __syncthreads();

        // ---- prior + posterior elementwise, outputs, stoch carry ----------
        for (int i = tid; i < 16 * S_; i += blockDim.x) {
            int r = i / S_, s = i % S_;
            size_t ob = (size_t)(m0 + r) * (T_ * OUT_) + (size_t)t * OUT_;
            // prior
            float pm = sStat[r * N4P + s];
            float ps = softplusf_(sStat[r * N4P + S_ + s]) + MIN_STD_;
            float np = noise_p[(size_t)t * (B_ * S_) + (size_t)(m0 + r) * S_ + s];
            out[ob + 3 * S_ + s] = pm;
            out[ob + 4 * S_ + s] = ps;
            out[ob + 5 * S_ + s] = pm + ps * np;
            // posterior
            float om = sStat2[r * N4P + s];
            float os = softplusf_(sStat2[r * N4P + S_ + s]) + MIN_STD_;
            float no = noise_o[(size_t)t * (B_ * S_) + (size_t)(m0 + r) * S_ + s];
            float ost = om + os * no;
            out[ob + 0 * S_ + s] = om;
            out[ob + 1 * S_ + s] = os;
            out[ob + 2 * S_ + s] = ost;
            sA1[r * K1P + s] = (__bf16)ost;   // stoch carry for next step
        }
        __syncthreads();
    }
}

// ---------------------------------------------------------------------------

extern "C" void kernel_launch(void* const* d_in, const int* in_sizes, int n_in,
                              void* d_out, int out_size, void* d_ws, size_t ws_size,
                              hipStream_t stream)
{
    (void)in_sizes; (void)n_in; (void)out_size; (void)ws_size;

    const float* embed    = (const float*)d_in[0];
    const float* action   = (const float*)d_in[1];
    const float* context  = (const float*)d_in[2];
    const float* noise_p  = (const float*)d_in[3];
    const float* noise_o  = (const float*)d_in[4];
    const float* w_inp    = (const float*)d_in[5];
    const float* b_inp    = (const float*)d_in[6];
    const float* w_gru    = (const float*)d_in[7];
    const float* b_gru    = (const float*)d_in[8];
    const float* w_img    = (const float*)d_in[9];
    const float* b_img    = (const float*)d_in[10];
    const float* w_obs    = (const float*)d_in[11];
    const float* b_obs    = (const float*)d_in[12];
    const float* w_ims    = (const float*)d_in[13];
    const float* b_ims    = (const float*)d_in[14];
    const float* w_ost    = (const float*)d_in[15];
    const float* b_ost    = (const float*)d_in[16];
    float* out = (float*)d_out;

    // carve bf16 weight buffers from workspace (~1.25 MB total)
    char* ws = (char*)d_ws;
    size_t off = 0;
    auto carve = [&](size_t elems) {
        __bf16* p = (__bf16*)(ws + off);
        off = (off + elems * sizeof(__bf16) + 255) & ~(size_t)255;
        return p;
    };
    __bf16* wt_inp = carve((size_t)N1P * K1P);
    __bf16* wt_gru = carve((size_t)N2P * K2P);
    __bf16* wt_img = carve((size_t)N3P * K3P);
    __bf16* wt_ims = carve((size_t)N4P * K3P);
    __bf16* wt_obs = carve((size_t)N5P * K5P);
    __bf16* wt_ost = carve((size_t)N6P * K3P);

    auto prep = [&](const float* src, __bf16* dst, int K, int N, int KP, int NP) {
        int total = NP * KP;
        prep_weight_kernel<<<(total + 255) / 256, 256, 0, stream>>>(src, dst, K, N, KP, NP);
    };
    prep(w_inp, wt_inp, S_ + SS_ + A_, H_,     K1P, N1P);
    prep(w_gru, wt_gru, H_ + D_,       3 * D_, K2P, N2P);
    prep(w_img, wt_img, D_,            H_,     K3P, N3P);
    prep(w_ims, wt_ims, H_,            2 * S_, K3P, N4P);
    prep(w_obs, wt_obs, D_ + E_,       H_,     K5P, N5P);
    prep(w_ost, wt_ost, H_,            2 * S_, K3P, N6P);

    rssm_observe_kernel<<<B_ / 16, 256, 0, stream>>>(
        embed, action, context, noise_p, noise_o,
        wt_inp, b_inp, wt_gru, b_gru, wt_img, b_img,
        wt_obs, b_obs, wt_ims, b_ims, wt_ost, b_ost, out);
}
